// UniDirectionalRNNTagger_76218489635004
// MI455X (gfx1250) — matrix operand order repaired and checked by reading, not compiled, not run on hardware
//
#include <hip/hip_runtime.h>
#include <cmath>

typedef __attribute__((ext_vector_type(16))) _Float16 v16h;
typedef __attribute__((ext_vector_type(8)))  _Float16 v8h;
typedef __attribute__((ext_vector_type(8)))  float    v8f;
typedef __attribute__((ext_vector_type(4)))  float    v4f;

#define VOCAB  50000
#define EMBED  256
#define HIDDEN 512
#define BATCH  64
#define SEQ    512
#define HP     520   // padded LDS h row stride (halves): 1040B -> distinct banks, 16B aligned
#define XP     520   // padded LDS x row stride (halves)

// workspace layout (bytes)
#define XPROJ_OFF 0ull                                         // f16 [B*S][H]   32 MB
#define WIH_OFF   (XPROJ_OFF + (size_t)BATCH*SEQ*HIDDEN*2)     // f16 [H][E]    256 KB
#define WHH_OFF   (WIH_OFF   + (size_t)HIDDEN*EMBED*2)         // f16 [H][H]    512 KB
#define FEAT_OFF  (WHH_OFF   + (size_t)HIDDEN*HIDDEN*2)        // f16 [B*S][H]   32 MB

// ---------------------------------------------------------------------------
// Kernel 0: convert f32 weights -> f16 once (W_ih, W_hh)
// ---------------------------------------------------------------------------
__global__ __launch_bounds__(256) void convert_weights(const float* __restrict__ wih,
                                                       const float* __restrict__ whh,
                                                       _Float16* __restrict__ wih_h,
                                                       _Float16* __restrict__ whh_h) {
    int i = blockIdx.x * 256 + threadIdx.x;
    if (i < HIDDEN * EMBED)  wih_h[i] = (_Float16)wih[i];
    if (i < HIDDEN * HIDDEN) whh_h[i] = (_Float16)whh[i];
}

// ---------------------------------------------------------------------------
// Kernel 1: x_proj[m][n] = emb[tok[m]] . W_ih[n] + b_ih[n] + b_hh[n]  (f16 out)
// One block per M-tile (16 rows), 8 waves x 64 output columns each.
// ---------------------------------------------------------------------------
__global__ __launch_bounds__(256) void xproj_kernel(const int*   __restrict__ tok,
                                                    const float* __restrict__ emb,
                                                    const _Float16* __restrict__ wih_h,
                                                    const float* __restrict__ b_ih,
                                                    const float* __restrict__ b_hh,
                                                    _Float16* __restrict__ xproj) {
    const int tid  = threadIdx.x;
    const int lane = tid & 31;
    const int wave = tid >> 5;
    const int l16  = lane >> 4;       // 0 or 1 (upper half-wave)
    const int lm   = lane & 15;
    const int m0   = blockIdx.x * 16;
    const int nb   = wave * 64;

    const float* erow = emb + (size_t)tok[m0 + lm] * EMBED;

    v8f acc[4] = {v8f{}, v8f{}, v8f{}, v8f{}};

    for (int k0 = 0; k0 < EMBED; k0 += 32) {
        // A tile 16x32: per lane one contiguous 16-element run (f32 -> f16)
        const int offa = k0 + (l16 << 4);
        v4f a0 = *(const v4f*)(erow + offa);
        v4f a1 = *(const v4f*)(erow + offa + 4);
        v4f a2 = *(const v4f*)(erow + offa + 8);
        v4f a3 = *(const v4f*)(erow + offa + 12);
        v16h a;
#pragma unroll
        for (int i = 0; i < 4; ++i) {
            a[i]      = (_Float16)a0[i];
            a[4 + i]  = (_Float16)a1[i];
            a[8 + i]  = (_Float16)a2[i];
            a[12 + i] = (_Float16)a3[i];
        }
        // B tiles 32x16: per lane one contiguous 16-half run of W_ih row n
        const int offb = k0 + (l16 << 4);
#pragma unroll
        for (int j = 0; j < 4; ++j) {
            const _Float16* brow = wih_h + (size_t)(nb + 16 * j + lm) * EMBED + offb;
            v8h b0 = *(const v8h*)(brow);
            v8h b1 = *(const v8h*)(brow + 8);
            v16h b;
#pragma unroll
            for (int i = 0; i < 8; ++i) { b[i] = b0[i]; b[8 + i] = b1[i]; }
            acc[j] = __builtin_amdgcn_wmma_f32_16x16x32_f16(false, a, false, b,
                                                            (short)0, acc[j], false, false);
        }
    }

#pragma unroll
    for (int j = 0; j < 4; ++j) {
        const int n = nb + 16 * j + lm;
        const float bias = b_ih[n] + b_hh[n];
#pragma unroll
        for (int r = 0; r < 8; ++r) {
            const int mloc = r + (l16 << 3);
            xproj[(size_t)(m0 + mloc) * HIDDEN + n] = (_Float16)(acc[j][r] + bias);
        }
    }
}

// ---------------------------------------------------------------------------
// Kernel 2: recurrence. 4 persistent WGs (16 batch rows each), 16 waves/WG.
//  - h kept in LDS (f16)
//  - W_hh streamed from L2 (f16), software-pipelined (double-buffered tiles)
//  - next timestep's x slice prefetched into LDS via global_load_async_to_lds
// ---------------------------------------------------------------------------
__global__ __launch_bounds__(512) void rnn_kernel(const _Float16* __restrict__ xproj,
                                                  const _Float16* __restrict__ whh_h,
                                                  _Float16* __restrict__ feat) {
    __shared__ _Float16 hbuf[16 * HP];
    __shared__ _Float16 xbuf[2][16 * XP];

    const int tid  = threadIdx.x;
    const int lane = tid & 31;
    const int wave = tid >> 5;          // 0..15, owns 32 output columns
    const int l16  = lane >> 4;
    const int lm   = lane & 15;
    const int g    = blockIdx.x;        // batch group 0..3
    const int n0   = wave * 32;

    for (int i = tid; i < 16 * HP; i += 512) hbuf[i] = (_Float16)0.f;

    const _Float16* brow0 = whh_h + (size_t)(n0 + lm) * HIDDEN;
    const _Float16* brow1 = whh_h + (size_t)(n0 + 16 + lm) * HIDDEN;
    const _Float16* arow  = hbuf + lm * HP;

    // warm this wave's W_hh slice into cache (global_prefetch_b8)
#pragma unroll
    for (int k = 0; k < HIDDEN; k += 64) {
        __builtin_prefetch(brow0 + k, 0, 1);
        __builtin_prefetch(brow1 + k, 0, 1);
    }

    // async prefetch of x slice for timestep tt into xbuf[buf]:
    // 1024 b128 chunks (16 rows x 64 chunks), 2 per thread, ASYNCcnt-tracked.
    auto issue_x = [&](int tt, int buf) {
#pragma unroll
        for (int c = 0; c < 2; ++c) {
            const int i    = c * 512 + tid;
            const int mloc = i >> 6;
            const int cin  = i & 63;
            const _Float16* gp = xproj +
                ((size_t)(g * 16 + mloc) * SEQ + tt) * HIDDEN + cin * 8;
            unsigned long long ga = (unsigned long long)(uintptr_t)gp;
            unsigned lp = (unsigned)(uintptr_t)&xbuf[buf][mloc * XP + cin * 8];
            asm volatile("global_load_async_to_lds_b128 %0, %1, off"
                         :: "v"(lp), "v"(ga) : "memory");
        }
    };

    issue_x(0, 0);
    __syncthreads();

    for (int t = 0; t < SEQ; ++t) {
        // prefetch next timestep (clamped duplicate on last iter, never read)
        const int tn = (t + 1 < SEQ) ? (t + 1) : (SEQ - 1);
        issue_x(tn, (t + 1) & 1);

        v8f acc0 = v8f{};
        v8f acc1 = v8f{};

        // software-pipelined GEMM over K: prefetch k+32 B tiles before use
        const int offb0 = (l16 << 4);
        v8h b0lo = *(const v8h*)(brow0 + offb0);
        v8h b0hi = *(const v8h*)(brow0 + offb0 + 8);
        v8h b1lo = *(const v8h*)(brow1 + offb0);
        v8h b1hi = *(const v8h*)(brow1 + offb0 + 8);

#pragma unroll
        for (int k0 = 0; k0 < HIDDEN; k0 += 32) {
            v8h nb0lo, nb0hi, nb1lo, nb1hi;
            if (k0 + 32 < HIDDEN) {
                const int offb = k0 + 32 + (l16 << 4);
                nb0lo = *(const v8h*)(brow0 + offb);
                nb0hi = *(const v8h*)(brow0 + offb + 8);
                nb1lo = *(const v8h*)(brow1 + offb);
                nb1hi = *(const v8h*)(brow1 + offb + 8);
            }
            const int offa = k0 + (l16 << 4);
            v8h alo = *(const v8h*)(arow + offa);        // ds_load_b128
            v8h ahi = *(const v8h*)(arow + offa + 8);
            v16h a, b0, b1;
#pragma unroll
            for (int i = 0; i < 8; ++i) {
                a[i] = alo[i];   a[8 + i] = ahi[i];
                b0[i] = b0lo[i]; b0[8 + i] = b0hi[i];
                b1[i] = b1lo[i]; b1[8 + i] = b1hi[i];
            }
            acc0 = __builtin_amdgcn_wmma_f32_16x16x32_f16(false, a, false, b0,
                                                          (short)0, acc0, false, false);
            acc1 = __builtin_amdgcn_wmma_f32_16x16x32_f16(false, a, false, b1,
                                                          (short)0, acc1, false, false);
            b0lo = nb0lo; b0hi = nb0hi; b1lo = nb1lo; b1hi = nb1hi;
        }

        // current step's x slice landed (in-order: only the 2 newest may remain)
        asm volatile("s_wait_asynccnt 0x2" ::: "memory");
        __syncthreads();   // all waves done reading old h; x[t] visible to all

        const _Float16* xs = &xbuf[t & 1][0];
#pragma unroll
        for (int j = 0; j < 2; ++j) {
            const v8f& acc = j ? acc1 : acc0;
            const int n = n0 + 16 * j + lm;
#pragma unroll
            for (int r = 0; r < 8; ++r) {
                const int mloc = r + (l16 << 3);
                const float v = tanhf((float)xs[mloc * XP + n] + acc[r]);
                hbuf[mloc * HP + n] = (_Float16)v;
                feat[((size_t)(g * 16 + mloc) * SEQ + t) * HIDDEN + n] = (_Float16)v;
            }
        }
        __syncthreads();   // new h visible before next step's A reads
    }
}

// ---------------------------------------------------------------------------
// Kernel 3: logits[m][c] = feat[m][:] . W_cls[c][:] + b_cls[c]   (c = 0,1)
// ---------------------------------------------------------------------------
__global__ __launch_bounds__(256) void cls_kernel(const _Float16* __restrict__ feat,
                                                  const float* __restrict__ wcls,
                                                  const float* __restrict__ bcls,
                                                  float* __restrict__ out) {
    const int wave = threadIdx.x >> 5;
    const int lane = threadIdx.x & 31;
    const size_t row = (size_t)blockIdx.x * 8 + wave;

    const _Float16* f = feat + row * HIDDEN + lane * 16;
    v8h f0 = *(const v8h*)(f);
    v8h f1 = *(const v8h*)(f + 8);
    const float* w0 = wcls + lane * 16;
    const float* w1 = wcls + HIDDEN + lane * 16;

    float s0 = 0.f, s1 = 0.f;
#pragma unroll
    for (int i = 0; i < 8; ++i) {
        s0 += (float)f0[i] * w0[i] + (float)f1[i] * w0[8 + i];
        s1 += (float)f0[i] * w1[i] + (float)f1[i] * w1[8 + i];
    }
#pragma unroll
    for (int off = 16; off > 0; off >>= 1) {
        s0 += __shfl_xor(s0, off);
        s1 += __shfl_xor(s1, off);
    }
    if (lane == 0) {
        out[row * 2 + 0] = s0 + bcls[0];
        out[row * 2 + 1] = s1 + bcls[1];
    }
}

// ---------------------------------------------------------------------------
extern "C" void kernel_launch(void* const* d_in, const int* in_sizes, int n_in,
                              void* d_out, int out_size, void* d_ws, size_t ws_size,
                              hipStream_t stream) {
    (void)in_sizes; (void)n_in; (void)out_size; (void)ws_size;

    const int*   tok  = (const int*)  d_in[0];
    const float* emb  = (const float*)d_in[1];
    const float* wih  = (const float*)d_in[2];
    const float* whh  = (const float*)d_in[3];
    const float* bih  = (const float*)d_in[4];
    const float* bhh  = (const float*)d_in[5];
    const float* wcls = (const float*)d_in[6];
    const float* bcls = (const float*)d_in[7];
    float* out = (float*)d_out;

    char* ws = (char*)d_ws;
    _Float16* xproj = (_Float16*)(ws + XPROJ_OFF);
    _Float16* wih_h = (_Float16*)(ws + WIH_OFF);
    _Float16* whh_h = (_Float16*)(ws + WHH_OFF);
    _Float16* feat  = (_Float16*)(ws + FEAT_OFF);

    convert_weights<<<(HIDDEN * HIDDEN + 255) / 256, 256, 0, stream>>>(wih, whh, wih_h, whh_h);
    xproj_kernel<<<(BATCH * SEQ) / 16, 256, 0, stream>>>(tok, emb, wih_h, bih, bhh, xproj);
    rnn_kernel<<<4, 512, 0, stream>>>(xproj, whh_h, feat);
    cls_kernel<<<(BATCH * SEQ) / 8, 256, 0, stream>>>(feat, wcls, bcls, out);
}
